// LSTM_74474732913122
// MI455X (gfx1250) — matrix operand order repaired and checked by reading, not compiled, not run on hardware
//
#include <hip/hip_runtime.h>
#include <hip/hip_bf16.h>

#define BB 128
#define TT 1024
#define DD 128
#define HH 256
#define H4 1024
#define CC 128

typedef __attribute__((ext_vector_type(16))) __bf16 bf16x16;
typedef __attribute__((ext_vector_type(8)))  float  f32x8;
typedef __attribute__((ext_vector_type(4)))  int    v4i;

union TileAB { bf16x16 v; uint4 u[2]; };

#ifndef __has_builtin
#define __has_builtin(x) 0
#endif
#if __has_builtin(__builtin_amdgcn_global_load_async_to_lds_b128)
#define HAVE_ASYNC_LDS 1
#else
#define HAVE_ASYNC_LDS 0
#endif

typedef __attribute__((address_space(1))) v4i g_v4i;   // global-memory 16B chunk
typedef __attribute__((address_space(3))) v4i l_v4i;   // LDS 16B chunk

// Async 16-byte global->LDS copy (ASYNCcnt path); sync fallback if builtin absent.
__device__ __forceinline__ void g2l_b128(void* lds, const void* g) {
#if HAVE_ASYNC_LDS
  __builtin_amdgcn_global_load_async_to_lds_b128((g_v4i*)g, (l_v4i*)lds, 0, 0);
#else
  *(uint4*)lds = *(const uint4*)g;
#endif
}
__device__ __forceinline__ void wait_async0() {
#if HAVE_ASYNC_LDS
#if __has_builtin(__builtin_amdgcn_s_wait_asynccnt)
  __builtin_amdgcn_s_wait_asynccnt(0);
#else
  asm volatile("s_wait_asynccnt 0x0" ::: "memory");
#endif
#endif
}

__device__ __forceinline__ unsigned short f32_to_bf16_rne(float f) {
  unsigned u = __builtin_bit_cast(unsigned, f);
  u += 0x7FFFu + ((u >> 16) & 1u);
  return (unsigned short)(u >> 16);
}
__device__ __forceinline__ float bf16_to_f32(unsigned short s) {
  unsigned u = ((unsigned)s) << 16;
  return __builtin_bit_cast(float, u);
}
__device__ __forceinline__ float fast_sigmoid(float x) {
  return 1.0f / (1.0f + __expf(-x));
}
__device__ __forceinline__ float fast_tanh(float x) {
  return 2.0f * fast_sigmoid(2.0f * x) - 1.0f;
}

// ---------------------------------------------------------------------------
// Kernel 1: repack weights to bf16, transposed+concatenated [N][K] layouts.
// ---------------------------------------------------------------------------
__global__ void __launch_bounds__(256) prep_weights(
    const float* wgx, const float* wgh, const float* bg,
    const float* wix, const float* wih, const float* bi,
    const float* wfx, const float* wfh, const float* bf,
    const float* wox, const float* woh, const float* bo,
    const float* wph,
    unsigned short* whT, unsigned short* wxT, unsigned short* wpT, float* bias)
{
  int n = blockIdx.x;          // 0..1023 (concat column: gate*256 + col)
  int k = threadIdx.x;         // 0..255
  int gate = n >> 8, col = n & 255;
  const float* Wh = (gate == 0) ? wgh : (gate == 1) ? wih : (gate == 2) ? wfh : woh;
  const float* Wx = (gate == 0) ? wgx : (gate == 1) ? wix : (gate == 2) ? wfx : wox;
  const float* Bv = (gate == 0) ? bg  : (gate == 1) ? bi  : (gate == 2) ? bf  : bo;
  whT[n * HH + k] = f32_to_bf16_rne(Wh[k * HH + col]);
  if (k < DD) wxT[n * DD + k] = f32_to_bf16_rne(Wx[k * HH + col]);
  if (n < CC) wpT[n * HH + k] = f32_to_bf16_rne(wph[k * CC + n]);
  if (k == 0) bias[n] = Bv[col];
}

// ---------------------------------------------------------------------------
// Kernel 2: convert x (fp32, [B*T,128]) to bf16, 8 elements/thread.
// ---------------------------------------------------------------------------
__global__ void __launch_bounds__(256) conv_x(const float* x, unsigned short* xb) {
  long i = (long)(blockIdx.x * blockDim.x + threadIdx.x) * 8;
  float4 a = *(const float4*)(x + i);
  float4 b = *(const float4*)(x + i + 4);
  union { unsigned short s[8]; uint4 u; } P;
  P.s[0] = f32_to_bf16_rne(a.x); P.s[1] = f32_to_bf16_rne(a.y);
  P.s[2] = f32_to_bf16_rne(a.z); P.s[3] = f32_to_bf16_rne(a.w);
  P.s[4] = f32_to_bf16_rne(b.x); P.s[5] = f32_to_bf16_rne(b.y);
  P.s[6] = f32_to_bf16_rne(b.z); P.s[7] = f32_to_bf16_rne(b.w);
  *(uint4*)(xb + i) = P.u;
}

// ---------------------------------------------------------------------------
// Kernel 3: xproj GEMM.  [B*T,128] x [128,1024] -> bf16 xproj stored [T,B,1024].
// ---------------------------------------------------------------------------
__global__ void __launch_bounds__(256) xproj_gemm(
    const unsigned short* __restrict__ xb, const unsigned short* __restrict__ wxT,
    const float* __restrict__ bias, unsigned short* __restrict__ xproj)
{
  int m0   = blockIdx.x * 16;
  int wave = threadIdx.x >> 5;
  int lane = threadIdx.x & 31;
  int lrow = lane & 15;
  int hi   = lane >> 4;

  TileAB A[4];
  const unsigned short* xrow = xb + (long)(m0 + lrow) * DD;
  #pragma unroll
  for (int kt = 0; kt < 4; ++kt) {
    int c0 = kt * 32 + hi * 16;
    A[kt].u[0] = *(const uint4*)(xrow + c0);
    A[kt].u[1] = *(const uint4*)(xrow + c0 + 8);
  }

  int bidx  = m0 >> 10;
  int tbase = m0 & 1023;

  #pragma unroll
  for (int j = 0; j < 8; ++j) {
    int n0   = (wave * 8 + j) * 16;
    int ncol = n0 + lrow;
    const unsigned short* wrow = wxT + (long)ncol * DD;
    f32x8 acc = {0.f,0.f,0.f,0.f,0.f,0.f,0.f,0.f};
    #pragma unroll
    for (int kt = 0; kt < 4; ++kt) {
      TileAB Bt;
      int c0 = kt * 32 + hi * 16;
      Bt.u[0] = *(const uint4*)(wrow + c0);
      Bt.u[1] = *(const uint4*)(wrow + c0 + 8);
      acc = __builtin_amdgcn_wmma_f32_16x16x32_bf16(
          false, A[kt].v, false, Bt.v, (short)0, acc, false, false);
    }
    float bv = bias[ncol];
    #pragma unroll
    for (int e = 0; e < 8; ++e) {
      int t = tbase + e + hi * 8;
      long oidx = ((long)t * BB + bidx) * H4 + ncol;
      xproj[oidx] = f32_to_bf16_rne(acc[e] + bv);
    }
  }
}

// ---------------------------------------------------------------------------
// Kernel 4: persistent LSTM scan + final projection.
// 8 WGs, each owns 16 independent batch rows (no cross-WG sync).
// LDS: w_h^T K 0..95 pinned (208KB) | h double buffer | xproj stage double
// buffer (2x32KB) filled by async global->LDS copies one step ahead.
// ---------------------------------------------------------------------------
__global__ void __launch_bounds__(256) lstm_scan(
    const unsigned short* __restrict__ xproj, const unsigned short* __restrict__ whT,
    const unsigned short* __restrict__ wpT, const float* __restrict__ bp,
    float* __restrict__ out)
{
  extern __shared__ char smem[];
  const int WH_STRIDE = 104;   // 96 data + 8 pad elements (208B rows)
  const int H_STRIDE  = 264;   // 256 + 8 pad elements (528B rows)
  unsigned short* whL  = (unsigned short*)smem;                              // [1024][104]
  unsigned short* hbuf = (unsigned short*)(smem + 1024 * WH_STRIDE * 2);     // 2x[16][264]
  unsigned short* xpL  = (unsigned short*)(smem + 1024 * WH_STRIDE * 2
                                                + 2 * 16 * H_STRIDE * 2);    // 2x[16][1024]

  int tid  = threadIdx.x;
  int wave = tid >> 5, lane = tid & 31, lrow = lane & 15, hi = lane >> 4;
  int m0   = blockIdx.x * 16;   // batch-row block

  // Pin K 0..95 of w_h^T into LDS.
  for (int n = tid; n < 1024; n += 256) {
    const uint4* src = (const uint4*)(whT + (long)n * HH);
    uint4* dst = (uint4*)(whL + n * WH_STRIDE);
    #pragma unroll
    for (int q = 0; q < 12; ++q) dst[q] = src[q];
  }
  // Zero h buffers (h0 = 0).
  for (int q = tid; q < 2 * 16 * H_STRIDE / 8; q += 256)
    ((uint4*)hbuf)[q] = make_uint4(0u, 0u, 0u, 0u);
  // Stage xproj slice for t=0 (async): 32KB contiguous -> xpL buf 0.
  {
    const unsigned short* gsrc = xproj + (long)m0 * H4;   // t=0
    #pragma unroll
    for (int q = 0; q < 8; ++q) {
      int off = (tid * 8 + q) * 8;                        // elements (16B chunks)
      g2l_b128(xpL + off, gsrc + off);
    }
  }
  wait_async0();
  __syncthreads();

  // Cell state in registers; wave owns h-col blocks {2w, 2w+1}.
  f32x8 c[2];
  #pragma unroll
  for (int b2 = 0; b2 < 2; ++b2)
    #pragma unroll
    for (int e = 0; e < 8; ++e) c[b2][e] = 0.0f;

  for (int t = 0; t < TT; ++t) {
    unsigned short* hin   = hbuf + (t & 1) * 16 * H_STRIDE;
    unsigned short* hout  = hbuf + ((t + 1) & 1) * 16 * H_STRIDE;
    unsigned short* xpcur = xpL + (t & 1) * 16 * H4;

    // Seed accumulators from LDS-staged xproj (bias already folded in).
    f32x8 acc[2][4];
    #pragma unroll
    for (int b2 = 0; b2 < 2; ++b2)
      #pragma unroll
      for (int g = 0; g < 4; ++g) {
        int ncol = g * 256 + (2 * wave + b2) * 16 + lrow;
        f32x8 a;
        #pragma unroll
        for (int e = 0; e < 8; ++e)
          a[e] = bf16_to_f32(xpcur[(e + hi * 8) * H4 + ncol]);
        acc[b2][g] = a;
      }

    // Kick off async staging of next step's xproj slice into the other buffer.
    if (t + 1 < TT) {
      const unsigned short* gsrc = xproj + ((long)(t + 1) * BB + m0) * H4;
      unsigned short* ldst = xpL + ((t + 1) & 1) * 16 * H4;
      #pragma unroll
      for (int q = 0; q < 8; ++q) {
        int off = (tid * 8 + q) * 8;
        g2l_b128(ldst + off, gsrc + off);
      }
    }

    // gates += h @ w_h  (kt 0..2 from LDS-resident w_h, kt 3..7 streamed from L2)
    #pragma unroll
    for (int kt = 0; kt < 8; ++kt) {
      TileAB At;
      int ac0 = kt * 32 + hi * 16;
      const unsigned short* ap = hin + lrow * H_STRIDE + ac0;
      At.u[0] = *(const uint4*)(ap);
      At.u[1] = *(const uint4*)(ap + 8);
      #pragma unroll
      for (int b2 = 0; b2 < 2; ++b2)
        #pragma unroll
        for (int g = 0; g < 4; ++g) {
          int nrow = g * 256 + (2 * wave + b2) * 16 + lrow;
          TileAB Bt;
          if (kt < 3) {
            const unsigned short* bptr = whL + nrow * WH_STRIDE + kt * 32 + hi * 16;
            Bt.u[0] = *(const uint4*)(bptr);
            Bt.u[1] = *(const uint4*)(bptr + 8);
          } else {
            const unsigned short* bptr = whT + (long)nrow * HH + kt * 32 + hi * 16;
            Bt.u[0] = *(const uint4*)(bptr);
            Bt.u[1] = *(const uint4*)(bptr + 8);
          }
          acc[b2][g] = __builtin_amdgcn_wmma_f32_16x16x32_bf16(
              false, At.v, false, Bt.v, (short)0, acc[b2][g], false, false);
        }
    }

    // Elementwise gate math; write new h (bf16) to the other buffer.
    #pragma unroll
    for (int b2 = 0; b2 < 2; ++b2) {
      int colb = (2 * wave + b2) * 16 + lrow;
      #pragma unroll
      for (int e = 0; e < 8; ++e) {
        float gg = fast_tanh(acc[b2][0][e]);
        float ii = fast_sigmoid(acc[b2][1][e]);
        float ff = fast_sigmoid(acc[b2][2][e]);
        float oo = fast_sigmoid(acc[b2][3][e]);
        float cn = gg * ii + c[b2][e] * ff;
        c[b2][e] = cn;
        float hn = fast_tanh(cn) * oo;
        hout[(e + hi * 8) * H_STRIDE + colb] = f32_to_bf16_rne(hn);
      }
    }
    wait_async0();       // this wave's async LDS writes for t+1 are complete
    __syncthreads();     // all waves' h writes + stage writes visible
  }

  // Final projection: out = h_T @ w_ph + b_p.  8 waves x one 16-col block each.
  unsigned short* hfin = hbuf + (TT & 1) * 16 * H_STRIDE;  // TT even -> buf0
  int n0 = wave * 16;
  f32x8 pacc = {0.f,0.f,0.f,0.f,0.f,0.f,0.f,0.f};
  #pragma unroll
  for (int kt = 0; kt < 8; ++kt) {
    TileAB At;
    int ac0 = kt * 32 + hi * 16;
    const unsigned short* ap = hfin + lrow * H_STRIDE + ac0;
    At.u[0] = *(const uint4*)(ap);
    At.u[1] = *(const uint4*)(ap + 8);
    TileAB Bt;
    const unsigned short* wp = wpT + (long)(n0 + lrow) * HH + kt * 32 + hi * 16;
    Bt.u[0] = *(const uint4*)(wp);
    Bt.u[1] = *(const uint4*)(wp + 8);
    pacc = __builtin_amdgcn_wmma_f32_16x16x32_bf16(
        false, At.v, false, Bt.v, (short)0, pacc, false, false);
  }
  float bv = bp[n0 + lrow];
  #pragma unroll
  for (int e = 0; e < 8; ++e)
    out[(m0 + e + hi * 8) * CC + n0 + lrow] = pacc[e] + bv;
}

// ---------------------------------------------------------------------------
extern "C" void kernel_launch(void* const* d_in, const int* in_sizes, int n_in,
                              void* d_out, int out_size, void* d_ws, size_t ws_size,
                              hipStream_t stream) {
  const float* x   = (const float*)d_in[0];
  const float* wgx = (const float*)d_in[1];
  const float* wgh = (const float*)d_in[2];
  const float* bg  = (const float*)d_in[3];
  const float* wix = (const float*)d_in[4];
  const float* wih = (const float*)d_in[5];
  const float* bi  = (const float*)d_in[6];
  const float* wfx = (const float*)d_in[7];
  const float* wfh = (const float*)d_in[8];
  const float* bf_ = (const float*)d_in[9];
  const float* wox = (const float*)d_in[10];
  const float* woh = (const float*)d_in[11];
  const float* bo  = (const float*)d_in[12];
  const float* wph = (const float*)d_in[13];
  const float* bp  = (const float*)d_in[14];
  float* out = (float*)d_out;

  char* ws = (char*)d_ws;
  unsigned short* xproj = (unsigned short*)ws;                          // 256 MB
  size_t off = (size_t)TT * BB * H4 * 2;
  unsigned short* xb    = (unsigned short*)(ws + off);                  // 32 MB
  off += (size_t)BB * TT * DD * 2;
  unsigned short* whT   = (unsigned short*)(ws + off);                  // 512 KB
  unsigned short* wxT   = whT + 1024 * HH;                              // 256 KB
  unsigned short* wpT   = wxT + 1024 * DD;                              // 64 KB
  float*          bias  = (float*)(wpT + CC * HH);                      // 4 KB

  prep_weights<<<1024, 256, 0, stream>>>(wgx, wgh, bg, wix, wih, bi,
                                         wfx, wfh, bf_, wox, woh, bo,
                                         wph, whT, wxT, wpT, bias);
  conv_x<<<8192, 256, 0, stream>>>(x, xb);
  xproj_gemm<<<(BB * TT) / 16, 256, 0, stream>>>(xb, wxT, bias, xproj);
  size_t smem = (size_t)1024 * 104 * 2      // w_h^T K 0..95  (208 KB)
              + (size_t)2 * 16 * 264 * 2    // h double buffer (16.5 KB)
              + (size_t)2 * 16 * 1024 * 2;  // xproj stage double buffer (64 KB)
  lstm_scan<<<BB / 16, 256, smem, stream>>>(xproj, whT, wpT, bp, out);
}
